// ImprovedGNN_82429012345509
// MI455X (gfx1250) — matrix operand order repaired and checked by reading, not compiled, not run on hardware
//
#include <hip/hip_runtime.h>
#include <hip/hip_bf16.h>

typedef __attribute__((ext_vector_type(16))) _Float16 v16h;
typedef __attribute__((ext_vector_type(8)))  _Float16 v8h;
typedef __attribute__((ext_vector_type(8)))  float    v8f;

#define N_NODES 50000
#define N_EDGES 800000
#define HID 128
#define EDGE_DIM 16
#define KPAD 288   // 272 padded to multiple of 32

union AFrag { v16h v; v8h h[2]; };

// ---------------------------------------------------------------- utilities
__global__ void k_zero_f32(float* p, int n) {
    int i = blockIdx.x * blockDim.x + threadIdx.x;
    if (i < n) p[i] = 0.f;
}

__global__ void k_deg_edge(const int* __restrict__ dst, float* __restrict__ deg, int nE) {
    int i = blockIdx.x * blockDim.x + threadIdx.x;
    if (i < nE) atomicAdd(&deg[dst[i]], 1.0f);
}

__global__ void k_deg_rsqrt(float* d, int n) {
    int i = blockIdx.x * blockDim.x + threadIdx.x;
    if (i < n) d[i] = rsqrtf(d[i] + 1.0f);
}

// f32 -> f16 (optional relu)
__global__ void k_cvt_f16(const float* __restrict__ in, _Float16* __restrict__ out,
                          int n, int doRelu) {
    int i = blockIdx.x * blockDim.x + threadIdx.x;
    if (i < n) {
        float v = in[i];
        if (doRelu) v = fmaxf(v, 0.f);
        out[i] = (_Float16)v;
    }
}

// Build transposed padded f16 weight: Wt[n*KP + k] = (k<K) ? W[k*N + n] : 0
__global__ void k_prep_wt(const float* __restrict__ W, _Float16* __restrict__ Wt,
                          int K, int N, int KP) {
    int idx = blockIdx.x * blockDim.x + threadIdx.x;
    if (idx >= N * KP) return;
    int n = idx / KP, k = idx % KP;
    Wt[idx] = (k < K) ? (_Float16)W[(size_t)k * N + n] : (_Float16)0.f;
}

// ------------------------------------------------- node GEMM: C = Ah @ Wt^T
// Ah: [M x 128] f16 row-major; Wt: [128 x 128] f16, Wt[n*128+k]; C: [M x 128] f32
// block = 256 (8 waves); wave w -> col tile w; blockIdx.x -> 16-row tile
__global__ void k_gemm_node(const _Float16* __restrict__ Ah,
                            const _Float16* __restrict__ Wt,
                            float* __restrict__ C) {
    int wave = threadIdx.x >> 5;
    int lane = threadIdx.x & 31;
    int hsel = lane >> 4;
    int m    = lane & 15;
    int row  = blockIdx.x * 16 + m;
    int col  = wave * 16 + m;

    const _Float16* arow = Ah + (size_t)row * 128;
    const _Float16* brow = Wt + (size_t)col * 128;

    v8f acc = {};
#pragma unroll
    for (int ks = 0; ks < 128; ks += 32) {
        AFrag a, b;
        a.h[0] = *(const v8h*)(arow + ks + 8 * hsel);
        a.h[1] = *(const v8h*)(arow + ks + 16 + 8 * hsel);
        b.h[0] = *(const v8h*)(brow + ks + 8 * hsel);
        b.h[1] = *(const v8h*)(brow + ks + 16 + 8 * hsel);
        acc = __builtin_amdgcn_wmma_f32_16x16x32_f16(false, a.v, false, b.v,
                                                     (short)0, acc, false, false);
    }
#pragma unroll
    for (int i = 0; i < 8; ++i) {
        int r = blockIdx.x * 16 + i + 8 * hsel;
        C[(size_t)r * 128 + col] = acc[i];
    }
}

// ------------------------------------------ GCN aggregation (scatter + self)
__global__ void k_agg_init(const float* __restrict__ T, const float* __restrict__ dinv,
                           const float* __restrict__ bias, float* __restrict__ G) {
    int idx = blockIdx.x * blockDim.x + threadIdx.x;
    if (idx >= N_NODES * HID) return;
    int node = idx >> 7, c = idx & 127;
    float di = dinv[node];
    G[idx] = T[idx] * di * di + bias[c];
}

// one wave per edge; lane covers 4 channels via float4
__global__ void k_agg_edge(const float* __restrict__ T, const float* __restrict__ dinv,
                           const int* __restrict__ src, const int* __restrict__ dst,
                           float* __restrict__ G, int nE) {
    int e    = (blockIdx.x * blockDim.x + threadIdx.x) >> 5;
    int lane = threadIdx.x & 31;
    if (e >= nE) return;
    int s = src[e], d = dst[e];
    float w = dinv[s] * dinv[d];
    const float4 v = *((const float4*)(T + (size_t)s * 128) + lane);
    float* g = G + (size_t)d * 128 + lane * 4;
    atomicAdd(g + 0, v.x * w);
    atomicAdd(g + 1, v.y * w);
    atomicAdd(g + 2, v.z * w);
    atomicAdd(g + 3, v.w * w);
}

// ------------------------------------------------------------- edge MLP
// per-lane A-run loader over concat [h[src](128) | h[dst](128) | ea(16) | pad]
__device__ inline v8h load_a_run(const _Float16* hs, const _Float16* hd,
                                 const _Float16* ea, int kb) {
    if (kb < 128) return *(const v8h*)(hs + kb);
    if (kb < 256) return *(const v8h*)(hd + kb - 128);
    if (kb < 272) return *(const v8h*)(ea + kb - 256);
    v8h z = {};
    return z;
}

// out[e] = relu([h[s],h[d],ea] @ Wm1 + bm1) @ Wm2 + bm2
// Wave handles a 16-edge tile. A fragments (9 K-steps) are loaded ONCE into
// registers (72 VGPRs); Wm1t is staged once per block into LDS and read via
// ds_load_b128 inside the col-tile loop. Bias + ReLU + Wm2 dot fused.
__global__ void k_edge_mlp(const _Float16* __restrict__ Hh,   // [N x 128]
                           const _Float16* __restrict__ Eh,   // [E x 16]
                           const _Float16* __restrict__ Wt,   // [128 x KPAD]
                           const int* __restrict__ src, const int* __restrict__ dst,
                           const float* __restrict__ bm1,     // [128]
                           const float* __restrict__ wm2,     // [128]
                           const float* __restrict__ bm2,     // [1]
                           float* __restrict__ out) {
    __shared__ _Float16 ldsW[HID * KPAD];   // 73728 B

    // cooperative one-shot copy of Wm1t into LDS (b128 granularity)
    {
        const v8h* gsrc = (const v8h*)Wt;
        v8h*       ldst = (v8h*)ldsW;
        const int nvec = HID * KPAD / 8;    // 4608
        for (int i = threadIdx.x; i < nvec; i += blockDim.x) ldst[i] = gsrc[i];
    }
    __syncthreads();

    int wave = threadIdx.x >> 5;
    int lane = threadIdx.x & 31;
    int hsel = lane >> 4;
    int m    = lane & 15;
    int et   = blockIdx.x * 8 + wave;   // 16-edge tile
    int e    = et * 16 + m;

    int s = src[e], d = dst[e];
    const _Float16* hs = Hh + (size_t)s * 128;
    const _Float16* hd = Hh + (size_t)d * 128;
    const _Float16* ea = Eh + (size_t)e * 16;

    // load all 9 A fragments once (branches fold at compile time per ks)
    AFrag a[KPAD / 32];
#pragma unroll
    for (int t = 0; t < KPAD / 32; ++t) {
        int ks = t * 32;
        a[t].h[0] = load_a_run(hs, hd, ea, ks + 8 * hsel);
        a[t].h[1] = load_a_run(hs, hd, ea, ks + 16 + 8 * hsel);
    }

    float eacc[8];
#pragma unroll
    for (int i = 0; i < 8; ++i) eacc[i] = 0.f;

    for (int j = 0; j < 8; ++j) {                 // output col tile
        const _Float16* brow = ldsW + (size_t)(j * 16 + m) * KPAD;
        v8f acc = {};
#pragma unroll
        for (int t = 0; t < KPAD / 32; ++t) {     // 9 K-steps
            int ks = t * 32;
            AFrag b;
            b.h[0] = *(const v8h*)(brow + ks + 8 * hsel);
            b.h[1] = *(const v8h*)(brow + ks + 16 + 8 * hsel);
            acc = __builtin_amdgcn_wmma_f32_16x16x32_f16(false, a[t].v, false, b.v,
                                                         (short)0, acc, false, false);
        }
        // fuse bias + relu + second-layer weight; this lane owns column n = j*16+m
        float b1v = bm1[j * 16 + m];
        float w2  = wm2[j * 16 + m];
#pragma unroll
        for (int i = 0; i < 8; ++i)
            eacc[i] += fmaxf(acc[i] + b1v, 0.f) * w2;
    }
    // reduce across the 16 lanes of each half-group (column dimension)
#pragma unroll
    for (int off = 1; off < 16; off <<= 1) {
#pragma unroll
        for (int i = 0; i < 8; ++i)
            eacc[i] += __shfl_xor(eacc[i], off, 32);
    }
    if (m == 0) {
        float bb = bm2[0];
#pragma unroll
        for (int i = 0; i < 8; ++i)
            out[et * 16 + i + 8 * hsel] = eacc[i] + bb;
    }
}

// ---------------------------------------------------------------- launcher
extern "C" void kernel_launch(void* const* d_in, const int* in_sizes, int n_in,
                              void* d_out, int out_size, void* d_ws, size_t ws_size,
                              hipStream_t stream) {
    const float* x   = (const float*)d_in[0];
    const int*   ei  = (const int*)d_in[1];
    const float* eat = (const float*)d_in[2];
    const float* W1  = (const float*)d_in[3];
    const float* b1  = (const float*)d_in[4];
    const float* W2  = (const float*)d_in[5];
    const float* b2  = (const float*)d_in[6];
    const float* Wm1 = (const float*)d_in[7];
    const float* bm1 = (const float*)d_in[8];
    const float* Wm2 = (const float*)d_in[9];
    const float* bm2 = (const float*)d_in[10];
    float* out = (float*)d_out;

    const int* src = ei;
    const int* dst = ei + N_EDGES;

    // workspace carve-up (all 256B aligned)
    char* w = (char*)d_ws;
    size_t off = 0;
    auto carve = [&](size_t bytes) {
        void* p = w + off;
        off += (bytes + 255) & ~(size_t)255;
        return p;
    };
    float*     dinv = (float*)carve((size_t)N_NODES * 4);
    float*     T    = (float*)carve((size_t)N_NODES * HID * 4);
    float*     G    = (float*)carve((size_t)N_NODES * HID * 4);
    _Float16*  xh   = (_Float16*)carve((size_t)N_NODES * HID * 2);
    _Float16*  eah  = (_Float16*)carve((size_t)N_EDGES * EDGE_DIM * 2);
    _Float16*  W1t  = (_Float16*)carve((size_t)HID * HID * 2);
    _Float16*  W2t  = (_Float16*)carve((size_t)HID * HID * 2);
    _Float16*  Wm1t = (_Float16*)carve((size_t)HID * KPAD * 2);
    (void)ws_size; (void)n_in; (void)in_sizes; (void)out_size;

    const int B = 256;
    // degrees -> dinv
    k_zero_f32 <<<(N_NODES + B - 1) / B, B, 0, stream>>>(dinv, N_NODES);
    k_deg_edge <<<(N_EDGES + B - 1) / B, B, 0, stream>>>(dst, dinv, N_EDGES);
    k_deg_rsqrt<<<(N_NODES + B - 1) / B, B, 0, stream>>>(dinv, N_NODES);

    // f16 staging
    k_cvt_f16<<<(N_NODES * HID + B - 1) / B, B, 0, stream>>>(x, xh, N_NODES * HID, 0);
    k_cvt_f16<<<(N_EDGES * EDGE_DIM + B - 1) / B, B, 0, stream>>>(eat, eah,
                                                                  N_EDGES * EDGE_DIM, 0);
    k_prep_wt<<<(HID * HID + B - 1) / B, B, 0, stream>>>(W1, W1t, HID, HID, HID);
    k_prep_wt<<<(HID * HID + B - 1) / B, B, 0, stream>>>(W2, W2t, HID, HID, HID);
    k_prep_wt<<<(HID * KPAD + B - 1) / B, B, 0, stream>>>(Wm1, Wm1t, 2 * HID + EDGE_DIM,
                                                          HID, KPAD);

    // ---- GCN layer 1
    k_gemm_node<<<N_NODES / 16, B, 0, stream>>>(xh, W1t, T);
    k_agg_init <<<(N_NODES * HID + B - 1) / B, B, 0, stream>>>(T, dinv, b1, G);
    k_agg_edge <<<N_EDGES / 8, B, 0, stream>>>(T, dinv, src, dst, G, N_EDGES);
    k_cvt_f16  <<<(N_NODES * HID + B - 1) / B, B, 0, stream>>>(G, xh, N_NODES * HID, 1);

    // ---- GCN layer 2
    k_gemm_node<<<N_NODES / 16, B, 0, stream>>>(xh, W2t, T);
    k_agg_init <<<(N_NODES * HID + B - 1) / B, B, 0, stream>>>(T, dinv, b2, G);
    k_agg_edge <<<N_EDGES / 8, B, 0, stream>>>(T, dinv, src, dst, G, N_EDGES);
    k_cvt_f16  <<<(N_NODES * HID + B - 1) / B, B, 0, stream>>>(G, xh, N_NODES * HID, 0);

    // ---- fused edge MLP (WMMA K=288, A in regs, B in LDS) + relu + reduce
    k_edge_mlp<<<(N_EDGES / 16) / 8, B, 0, stream>>>(xh, eah, Wm1t, src, dst,
                                                     bm1, Wm2, bm2, out);
}
